// RNNModel_35854386987644
// MI455X (gfx1250) — matrix operand order repaired and checked by reading, not compiled, not run on hardware
//
#include <hip/hip_runtime.h>
#include <math.h>

// CDNA5 / gfx1250: wave32, WMMA 16x16x32 f16 -> f32
typedef __attribute__((ext_vector_type(16))) _Float16 v16h;
typedef __attribute__((ext_vector_type(8)))  float    v8f;

union V16H { v16h v; _Float16 e[16]; };
union V8F  { v8f  v; float     e[8];  };

#define LOG2E 1.4426950408889634f

// Branch-free activations on v_exp_f32 / v_rcp_f32 (TRANS ops co-execute with WMMA).
__device__ __forceinline__ float fast_sigmoid(float x) {
    // 1 / (1 + exp(-x));  exp(-x) = exp2(-x*log2e).  exp2 -> inf/0 saturates correctly.
    return __builtin_amdgcn_rcpf(1.0f + __builtin_amdgcn_exp2f(x * (-LOG2E)));
}
__device__ __forceinline__ float fast_tanh(float x) {
    // tanh(x) = 1 - 2/(1 + exp(2x));  exp(2x) = exp2(x*2*log2e).
    const float r = __builtin_amdgcn_rcpf(1.0f + __builtin_amdgcn_exp2f(x * (2.0f * LOG2E)));
    return fmaf(-2.0f, r, 1.0f);
}

// One wave32 handles 16 batch rows.
//   D(tile) = A(weights 16x32 f16) x B(activations 32x16 f16) + C  -> 16(gate rows) x 16(batch)
// Gate rows are PERMUTED into A so that lane n ends up holding, across the
// 6 tiles x 8 D-VGPRs, exactly (i,f,g,o) for hidden units 0..11 of batch n,
// and lane n+16 holds units 12..23.  Elementwise LSTM update is lane-local.
//
// perm: slot s = tile*8 + (M&7);  u = (s>>2) + (M>=8 ? 12 : 0);  gate = s&3;
//       global gate row = gate*24 + u        (i=0..23, f=24..47, g=48..71, o=72..95)
__global__ __launch_bounds__(128, 1)
void lstm_anomaly_scan(const float* __restrict__ x,
                       const float* __restrict__ W_ih,
                       const float* __restrict__ W_hh,
                       const float* __restrict__ b_ih,
                       const float* __restrict__ b_hh,
                       const float* __restrict__ W_out,
                       const float* __restrict__ b_out,
                       float* __restrict__ out,
                       int Bsz, int Tsz)
{
    constexpr int NIN = 8;   // input features
    constexpr int NH  = 24;  // hidden size

    const int lane = threadIdx.x & 31;
    const int wave = (blockIdx.x * blockDim.x + threadIdx.x) >> 5;
    const int b0   = wave * 16;
    const int m    = lane & 15;       // A row / D column index served by this lane
    const bool hi  = lane >= 16;
    const int bl   = b0 + m;          // batch row this lane serves

    // ---------- loop-invariant operands: A matrices (f16) + bias-as-C (f32) ----------
    V16H A_ih[6], A_hh[6];
    V8F  Cb[6];
    #pragma unroll
    for (int t6 = 0; t6 < 6; ++t6) {
        // A-matrix as filled here (contiguous k, B uses the same): lanes 0-15 hold row M=lane, K=0..15 (halves 0-15);
        // lanes 16-31 hold row M=lane-16, K=16..31 (halves 0-15).
        const int s   = t6 * 8 + (m & 7);
        const int u   = (s >> 2) + ((m >> 3) * 12);
        const int row = (s & 3) * NH + u;                 // permuted gate row
        #pragma unroll
        for (int k = 0; k < 16; ++k) { A_ih[t6].e[k] = (_Float16)0.0f; A_hh[t6].e[k] = (_Float16)0.0f; }
        if (!hi) {
            // K = 0..7 valid for W_ih (I=8); K = 0..15 for W_hh (H=24)
            #pragma unroll
            for (int k = 0; k < 8; ++k) A_ih[t6].e[k]     = (_Float16)W_ih[row * NIN + k];
            #pragma unroll
            for (int k = 0; k < 8; ++k) A_hh[t6].e[k]     = (_Float16)W_hh[row * NH + k];
            #pragma unroll
            for (int k = 0; k < 8; ++k) A_hh[t6].e[8 + k] = (_Float16)W_hh[row * NH + 8 + k];
        } else {
            // high-lane K range 16..23 valid only for W_hh
            #pragma unroll
            for (int k = 0; k < 8; ++k) A_hh[t6].e[k]     = (_Float16)W_hh[row * NH + 16 + k];
        }
        // C/D layout: VGPR r of this lane is row M = r (+8 for hi lanes) of the tile.
        #pragma unroll
        for (int r = 0; r < 8; ++r) {
            const int ss = t6 * 8 + r;
            const int uu = (ss >> 2) + (hi ? 12 : 0);
            const int rr = (ss & 3) * NH + uu;
            Cb[t6].e[r] = b_ih[rr] + b_hh[rr];
        }
    }

    // Output projection weights: lane n covers units 0..11, lane n+16 covers 12..23
    float wout[12];
    #pragma unroll
    for (int j = 0; j < 12; ++j) wout[j] = W_out[(hi ? 12 : 0) + j];
    const float bout = b_out[0];

    // LSTM state (12 hidden units per lane), pred carry
    float hreg[12], creg[12];
    #pragma unroll
    for (int j = 0; j < 12; ++j) { hreg[j] = 0.0f; creg[j] = 0.0f; }

    const float* xptr = x + (size_t)bl * Tsz * NIN;
    float pred = xptr[0];                       // pred0 = x[b,0,0]

    // B_h (h^T columns): lane n holds K=0..15 -> h[n][0..15]; lane n+16 holds K=16..31 -> h[n][16..23],0pad
    V16H Bh;
    #pragma unroll
    for (int k = 0; k < 16; ++k) Bh.e[k] = (_Float16)0.0f;

    for (int t = 0; t < Tsz; ++t) {
        const float4 f0 = *(const float4*)(xptr);       // feats 0..3
        const float4 f1 = *(const float4*)(xptr + 4);   // feats 4..7
        xptr += NIN;
        __builtin_prefetch(xptr + 16 * NIN, 0, 1);      // global_prefetch_b8, warm ahead

        // anomaly substitution on feature 0 (branch-free; only low lanes feed B_x)
        const float actual = f0.x;
        const float in0 = (fabsf(pred - actual) > 0.1f) ? pred : actual;

        V16H Bx;
        #pragma unroll
        for (int k = 0; k < 16; ++k) Bx.e[k] = (_Float16)0.0f;
        if (!hi) {
            Bx.e[0] = (_Float16)in0;  Bx.e[1] = (_Float16)f0.y;
            Bx.e[2] = (_Float16)f0.z; Bx.e[3] = (_Float16)f0.w;
            Bx.e[4] = (_Float16)f1.x; Bx.e[5] = (_Float16)f1.y;
            Bx.e[6] = (_Float16)f1.z; Bx.e[7] = (_Float16)f1.w;
        }

        // ---- 6 gate tiles: D = A_ih*Bx + bias, then accumulate A_hh*Bh; elementwise per tile ----
        float predAcc = 0.0f;
        #pragma unroll
        for (int t6 = 0; t6 < 6; ++t6) {
            v8f d = __builtin_amdgcn_wmma_f32_16x16x32_f16(
                        false, A_ih[t6].v, false, Bx.v, (short)0, Cb[t6].v, false, false);
            d = __builtin_amdgcn_wmma_f32_16x16x32_f16(
                        false, A_hh[t6].v, false, Bh.v, (short)0, d, false, false);
            V8F dd; dd.v = d;
            #pragma unroll
            for (int p = 0; p < 2; ++p) {
                const int u = 2 * t6 + p;               // local hidden unit
                const float ig = dd.e[4 * p + 0];
                const float fg = dd.e[4 * p + 1];
                const float gg = dd.e[4 * p + 2];
                const float og = dd.e[4 * p + 3];
                const float cn = fast_sigmoid(fg) * creg[u] + fast_sigmoid(ig) * fast_tanh(gg);
                const float hn = fast_sigmoid(og) * fast_tanh(cn);
                creg[u] = cn;
                hreg[u] = hn;
                predAcc = fmaf(hn, wout[u], predAcc);
            }
        }

        // 24->1 projection: lane n has units 0..11 partial, lane n+16 has 12..23
        const float predNew = predAcc + __shfl_xor(predAcc, 16, 32) + bout;
        pred = predNew;
        if (!hi) out[(size_t)bl * Tsz + t] = predNew;

        // rebuild B_h for the next step (shuffles outside divergent region)
        const float hp0 = __shfl_xor(hreg[0], 16, 32);
        const float hp1 = __shfl_xor(hreg[1], 16, 32);
        const float hp2 = __shfl_xor(hreg[2], 16, 32);
        const float hp3 = __shfl_xor(hreg[3], 16, 32);
        if (!hi) {
            #pragma unroll
            for (int k = 0; k < 12; ++k) Bh.e[k] = (_Float16)hreg[k];
            Bh.e[12] = (_Float16)hp0; Bh.e[13] = (_Float16)hp1;
            Bh.e[14] = (_Float16)hp2; Bh.e[15] = (_Float16)hp3;
        } else {
            #pragma unroll
            for (int k = 0; k < 8; ++k) Bh.e[k] = (_Float16)hreg[4 + k];   // h[16..23]
            #pragma unroll
            for (int k = 8; k < 16; ++k) Bh.e[k] = (_Float16)0.0f;         // K=24..31 pad
        }
    }
}

extern "C" void kernel_launch(void* const* d_in, const int* in_sizes, int n_in,
                              void* d_out, int out_size, void* d_ws, size_t ws_size,
                              hipStream_t stream)
{
    const float* x     = (const float*)d_in[0];
    const float* W_ih  = (const float*)d_in[1];
    const float* W_hh  = (const float*)d_in[2];
    const float* b_ih  = (const float*)d_in[3];
    const float* b_hh  = (const float*)d_in[4];
    const float* W_out = (const float*)d_in[5];
    const float* b_out = (const float*)d_in[6];
    float* out = (float*)d_out;

    const int T = 2048;
    int B = (out_size > 0) ? (out_size / T) : 2048;   // B*T*1 outputs
    if (B <= 0) B = 2048;

    const int block = 128;                 // 4 wave32 per block
    const int wavesTotal = B / 16;         // 16 batch rows per wave -> 128 waves
    const int grid = (wavesTotal * 32 + block - 1) / block;

    lstm_anomaly_scan<<<grid, block, 0, stream>>>(x, W_ih, W_hh, b_ih, b_hh,
                                                  W_out, b_out, out, B, T);
}
